// LSTMModel_19481971654943
// MI455X (gfx1250) — matrix operand order repaired and checked by reading, not compiled, not run on hardware
//
#include <hip/hip_runtime.h>
#include <hip/hip_bf16.h>

// ---------------------------------------------------------------------------
// Problem constants (from reference): B=1024, T=256, I=32, H=192, O=15
// ---------------------------------------------------------------------------
constexpr int B_ = 1024;
constexpr int T_ = 256;
constexpr int I_ = 32;
constexpr int H_ = 192;
constexpr int G_ = 4 * H_;      // 768 gate columns (i,f,g,o)
constexpr int O_ = 15;

constexpr int TILE_B  = 16;     // batch rows per workgroup (= WMMA M)
constexpr int WAVES   = H_ / 16;                 // 12 waves, one per 16 hidden units
constexpr int THREADS = WAVES * 32;              // 384 threads
constexpr int NTILES  = G_ / 16;                 // 48 N-tiles of 16 gate columns

typedef float  v8f   __attribute__((ext_vector_type(8)));
typedef __bf16 v16bf __attribute__((ext_vector_type(16)));

union AFrag { v16bf v; uint4 q[2]; };
union BFrag { v16bf v; uint4 q[2]; };

struct __align__(8) bf16x4 { __bf16 v[4]; };

// ---------------------------------------------------------------------------
// Weight packing: Wcat (k x 768), k in [0,H) -> W_hh^T rows, k in [H,H+IN) ->
// W_ih^T rows. Stored pre-swizzled in the exact CDNA5 B-matrix (32x16 bf16)
// register layout: fragment (kb, nt) is 512 contiguous bf16, ordered
// [lane][elem], so each lane fetches its 32 bytes with global_load_b128.
// B layout: lanes 0-15 hold K=0..7,16..23, lanes 16-31 hold K=8..15,24..31.
// ---------------------------------------------------------------------------
__global__ void pack_weights_kernel(const float* __restrict__ Whh,
                                    const float* __restrict__ Wih,
                                    const float* __restrict__ bih,
                                    const float* __restrict__ bhh,
                                    int IN, int KB,
                                    __bf16* __restrict__ Wsw,
                                    float*  __restrict__ bias) {
    const int total = KB * NTILES * 512;
    for (int e = blockIdx.x * blockDim.x + threadIdx.x; e < total;
         e += gridDim.x * blockDim.x) {
        const int frag = e >> 9;          // /512
        const int r    = e & 511;
        const int lane = r >> 4;
        const int el   = r & 15;
        const int kb   = frag / NTILES;
        const int nt   = frag % NTILES;
        const int n    = nt * 16 + (lane & 15);              // gate column
        const int k    = kb * 32 + (el & 7) + ((lane & 16) ? 8 : 0) + ((el & 8) ? 16 : 0);
        float w;
        if (k < H_) w = Whh[n * H_ + k];
        else        w = Wih[n * IN + (k - H_)];
        Wsw[e] = (__bf16)w;
    }
    for (int g = blockIdx.x * blockDim.x + threadIdx.x; g < G_;
         g += gridDim.x * blockDim.x)
        bias[g] = bih[g] + bhh[g];
}

// Branch-free fast activations (no ocml range-reduction branches on the
// sequential scan path).  exp overflow -> inf -> rcp -> 0 gives exact limits.
__device__ __forceinline__ float sigmoid_f(float x) {
    return __builtin_amdgcn_rcpf(1.0f + __expf(-x));
}
__device__ __forceinline__ float tanh_f(float x) {
    const float t = __expf(-2.0f * x);
    return fmaf(2.0f, __builtin_amdgcn_rcpf(1.0f + t), -1.0f);
}

// ---------------------------------------------------------------------------
// Persistent recurrent layer. One workgroup per 16-batch tile, loops over T.
// Wave w computes gate tiles {i,f,g,o} x hidden[16w,16w+16) via
// v_wmma_f32_16x16x32_bf16 with K = H + IN ([h_prev, x_t] concat in LDS).
// ---------------------------------------------------------------------------
template <int IN, bool FIRST>
__global__ __launch_bounds__(THREADS, 1)
void lstm_layer_kernel(const float*  __restrict__ xin,   // FIRST: [B,T,I] f32
                       const __bf16* __restrict__ hin,   // !FIRST: [B,T,H] bf16
                       const __bf16* __restrict__ Wsw,   // [KB][48][512] bf16
                       const float*  __restrict__ bias,  // [768] f32
                       __bf16*       __restrict__ hout,  // FIRST: [B,T,H] bf16
                       float*        __restrict__ hlast) // !FIRST: [B,H] f32
{
    constexpr int KTOT = H_ + IN;        // 224 (layer0) / 384 (layer1)
    constexpr int KB   = KTOT / 32;      // 7 / 12 K-steps

    __shared__ __align__(16) __bf16 cat[TILE_B][KTOT];   // [h_prev | x_t]

    const int tid   = threadIdx.x;
    const int lane  = tid & 31;
    const int wave  = tid >> 5;                 // 0..11  -> hidden chunk
    const int bBase = blockIdx.x * TILE_B;

    // C/D layout: lane holds column n = lane&15, rows m = 8*(lane>>4)+r.
    const int ncol  = lane & 15;
    const int mBase = (lane >> 4) << 3;
    const int koffL = (lane & 16) ? 8 : 0;      // A-layout lane-half K offset

    // Per-lane gate biases (same for all 8 rows of the tile column).
    float bval[4];
#pragma unroll
    for (int gt = 0; gt < 4; ++gt)
        bval[gt] = bias[gt * H_ + wave * 16 + ncol];

    // Cell state: 8 batch rows x this lane's hidden unit.
    float cst[8];
#pragma unroll
    for (int r = 0; r < 8; ++r) cst[r] = 0.0f;

    // h_prev = 0 at t = 0.
    for (int idx = tid; idx < TILE_B * H_; idx += THREADS)
        cat[idx / H_][idx % H_] = (__bf16)0.0f;

    // A-fragment loader: with koffL folded into the base, elems 0..7 and
    // 8..15 are two 16B-contiguous LDS chunks (two ds_load_b128).
    auto loadA = [&](int kb) {
        AFrag A;
        const __bf16* row = &cat[ncol][kb * 32 + koffL];
        A.q[0] = *reinterpret_cast<const uint4*>(row);
        A.q[1] = *reinterpret_cast<const uint4*>(row + 16);
        return A;
    };

    for (int t = 0; t < T_; ++t) {
        // ---- stage x_t into cat[:, H : H+IN] (vectorized) ---------------
        if constexpr (FIRST) {
            if (tid < TILE_B * (I_ / 4)) {               // 128 float4 loads
                const int m  = tid >> 3;                 // I_/4 == 8
                const int i4 = tid & 7;
                const float4 xv = *reinterpret_cast<const float4*>(
                    xin + ((size_t)(bBase + m) * T_ + t) * I_ + i4 * 4);
                bf16x4 b;
                b.v[0] = (__bf16)xv.x; b.v[1] = (__bf16)xv.y;
                b.v[2] = (__bf16)xv.z; b.v[3] = (__bf16)xv.w;
                *reinterpret_cast<bf16x4*>(&cat[m][H_ + i4 * 4]) = b;
            }
        } else {
#pragma unroll
            for (int it = 0; it < 2; ++it) {             // 768 bf16x4 loads
                const int idx = tid + it * THREADS;
                const int m = idx / (H_ / 4);
                const int q = idx % (H_ / 4);
                *reinterpret_cast<bf16x4*>(&cat[m][H_ + q * 4]) =
                    *reinterpret_cast<const bf16x4*>(
                        hin + ((size_t)(bBase + m) * T_ + t) * H_ + q * 4);
            }
        }
        __syncthreads();

        // ---- gate GEMM: [16 x KTOT] @ [KTOT x 64(gate cols)] -------------
        v8f acc[4];
#pragma unroll
        for (int gt = 0; gt < 4; ++gt) {
            v8f a;
#pragma unroll
            for (int r = 0; r < 8; ++r) a[r] = bval[gt];
            acc[gt] = a;
        }

        AFrag Acur = loadA(0);
#pragma unroll
        for (int kb = 0; kb < KB; ++kb) {
            AFrag Anext;
            if (kb + 1 < KB) Anext = loadA(kb + 1);      // pipelined A load
#pragma unroll
            for (int gt = 0; gt < 4; ++gt) {
                const int nt = gt * WAVES + wave;
                const __bf16* wb =
                    Wsw + ((size_t)(kb * NTILES + nt) * 512 + lane * 16);
                BFrag Bf;
                Bf.q[0] = *reinterpret_cast<const uint4*>(wb);
                Bf.q[1] = *reinterpret_cast<const uint4*>(wb + 8);
                acc[gt] = __builtin_amdgcn_wmma_f32_16x16x32_bf16(
                    false, Acur.v, false, Bf.v, (short)0, acc[gt], false, false);
            }
            if (kb + 1 < KB) Acur = Anext;
        }

        // ---- elementwise LSTM cell (i,f,g,o in the same lane) ------------
        const v8f vi = acc[0], vf = acc[1], vg = acc[2], vo = acc[3];
        __bf16 hb[8];
#pragma unroll
        for (int r = 0; r < 8; ++r) {
            const float i_s = sigmoid_f(vi[r]);
            const float f_s = sigmoid_f(vf[r]);
            const float g_t = tanh_f(vg[r]);
            const float o_s = sigmoid_f(vo[r]);
            const float cn  = f_s * cst[r] + i_s * g_t;
            cst[r] = cn;
            hb[r]  = (__bf16)(o_s * tanh_f(cn));
        }

        __syncthreads();   // all A reads done before overwriting h region

        const int hcol = wave * 16 + ncol;
#pragma unroll
        for (int r = 0; r < 8; ++r)
            cat[mBase + r][hcol] = hb[r];

        __syncthreads();   // h tile complete

        // ---- emit layer output (vectorized) ------------------------------
        if constexpr (FIRST) {
#pragma unroll
            for (int it = 0; it < 2; ++it) {
                const int idx = tid + it * THREADS;      // < 768
                const int m = idx / (H_ / 4);
                const int q = idx % (H_ / 4);
                *reinterpret_cast<bf16x4*>(
                    hout + ((size_t)(bBase + m) * T_ + t) * H_ + q * 4) =
                    *reinterpret_cast<const bf16x4*>(&cat[m][q * 4]);
            }
        } else {
            if (t == T_ - 1) {
                for (int idx = tid; idx < TILE_B * H_; idx += THREADS) {
                    const int m = idx / H_, h = idx % H_;
                    hlast[(size_t)(bBase + m) * H_ + h] = (float)cat[m][h];
                }
            }
        }
    }
}

// ---------------------------------------------------------------------------
// Final FC: [1024,192] @ [192,15]^T + b  (trivially small)
// ---------------------------------------------------------------------------
__global__ void fc_kernel(const float* __restrict__ hlast,
                          const float* __restrict__ Wfc,
                          const float* __restrict__ bfc,
                          float* __restrict__ out) {
    const int id = blockIdx.x * blockDim.x + threadIdx.x;
    if (id >= B_ * O_) return;
    const int b = id / O_, o = id % O_;
    const float4* hb = reinterpret_cast<const float4*>(hlast + (size_t)b * H_);
    const float4* wo = reinterpret_cast<const float4*>(Wfc + (size_t)o * H_);
    float s = bfc[o];
#pragma unroll 4
    for (int h4 = 0; h4 < H_ / 4; ++h4) {
        const float4 hv = hb[h4], wv = wo[h4];
        s = fmaf(hv.x, wv.x, s);
        s = fmaf(hv.y, wv.y, s);
        s = fmaf(hv.z, wv.z, s);
        s = fmaf(hv.w, wv.w, s);
    }
    out[id] = s;
}

// ---------------------------------------------------------------------------
// Launch
// ---------------------------------------------------------------------------
extern "C" void kernel_launch(void* const* d_in, const int* in_sizes, int n_in,
                              void* d_out, int out_size, void* d_ws,
                              size_t ws_size, hipStream_t stream) {
    (void)in_sizes; (void)n_in; (void)out_size; (void)ws_size;

    const float* x     = (const float*)d_in[0];
    const float* W_ih0 = (const float*)d_in[1];
    const float* W_hh0 = (const float*)d_in[2];
    const float* b_ih0 = (const float*)d_in[3];
    const float* b_hh0 = (const float*)d_in[4];
    const float* W_ih1 = (const float*)d_in[5];
    const float* W_hh1 = (const float*)d_in[6];
    const float* b_ih1 = (const float*)d_in[7];
    const float* b_hh1 = (const float*)d_in[8];
    const float* W_fc  = (const float*)d_in[9];
    const float* b_fc  = (const float*)d_in[10];
    float* out = (float*)d_out;

    // workspace layout (256B aligned)
    constexpr size_t SZ_WSW0 = (size_t)(H_ + I_) * G_ * 2;   // 344064
    constexpr size_t SZ_WSW1 = (size_t)(H_ + H_) * G_ * 2;   // 589824
    constexpr size_t SZ_BIAS = (size_t)G_ * 4;               // 3072
    constexpr size_t SZ_H1   = (size_t)B_ * T_ * H_ * 2;     // ~96 MB
    char* ws = (char*)d_ws;
    size_t off = 0;
    __bf16* Wsw0  = (__bf16*)(ws + off); off += SZ_WSW0; off = (off + 255) & ~(size_t)255;
    __bf16* Wsw1  = (__bf16*)(ws + off); off += SZ_WSW1; off = (off + 255) & ~(size_t)255;
    float*  bias0 = (float*) (ws + off); off += SZ_BIAS; off = (off + 255) & ~(size_t)255;
    float*  bias1 = (float*) (ws + off); off += SZ_BIAS; off = (off + 255) & ~(size_t)255;
    __bf16* h1    = (__bf16*)(ws + off); off += SZ_H1;   off = (off + 255) & ~(size_t)255;
    float*  hlast = (float*) (ws + off);

    constexpr int KB0 = (H_ + I_) / 32;   // 7
    constexpr int KB1 = (H_ + H_) / 32;   // 12

    pack_weights_kernel<<<256, 256, 0, stream>>>(W_hh0, W_ih0, b_ih0, b_hh0,
                                                 I_, KB0, Wsw0, bias0);
    pack_weights_kernel<<<256, 256, 0, stream>>>(W_hh1, W_ih1, b_ih1, b_hh1,
                                                 H_, KB1, Wsw1, bias1);

    lstm_layer_kernel<I_, true><<<B_ / TILE_B, THREADS, 0, stream>>>(
        x, nullptr, Wsw0, bias0, h1, nullptr);
    lstm_layer_kernel<H_, false><<<B_ / TILE_B, THREADS, 0, stream>>>(
        nullptr, h1, Wsw1, bias1, nullptr, hlast);

    fc_kernel<<<(B_ * O_ + 255) / 256, 256, 0, stream>>>(hlast, W_fc, b_fc, out);
}
